// VisionLanguageCrossAttention_65309272703202
// MI455X (gfx1250) — matrix operand order repaired and checked by reading, not compiled, not run on hardware
//
#include <hip/hip_runtime.h>

typedef __bf16 bf16_t;
typedef __attribute__((ext_vector_type(16))) __bf16 v16bf;
typedef __attribute__((ext_vector_type(8)))  float  v8f;

#define TILE_M 128
#define TILE_N 128
#define TILE_K 32
#define LDSS   40   // padded LDS row stride in bf16 elems: 80B rows, 16B aligned, conflict-free frag reads

static __device__ inline v8f zero_v8f() {
  v8f z;
#pragma unroll
  for (int i = 0; i < 8; ++i) z[i] = 0.0f;
  return z;
}

// ---------------------------------------------------------------------------
// Generic bf16 WMMA GEMM:  C[z] = alpha * A[z] @ B[z]^T + bias
//   A: [M,K] row-major (lda), per-batch offset (z/aDiv)*aOuter + (z%aDiv)*aInner
//   B: [N,K] row-major (ldb) (i.e. weights [out,in]), same offset scheme
//   C: f32 and/or bf16 outputs, [M,N] row-major (ldc)
// Requirements: K % 32 == 0; all A/B base offsets 16B aligned (true for all uses)
// ---------------------------------------------------------------------------
__global__ __launch_bounds__(256)
void k_gemm_bf16(const bf16_t* __restrict__ A, int lda, int aDiv, long aOuter, long aInner,
                 const bf16_t* __restrict__ Bw, int ldb, int bDiv, long bOuter, long bInner,
                 const float* __restrict__ bias,
                 float* __restrict__ Cf, bf16_t* __restrict__ Ch,
                 int ldc, int cDiv, long cOuter, long cInner,
                 int M, int N, int K, float alpha)
{
  __shared__ __align__(16) bf16_t As[TILE_M * LDSS];
  __shared__ __align__(16) bf16_t Bs[TILE_N * LDSS];

  const int z = blockIdx.z;
  const bf16_t* Ab = A  + (long)(z / aDiv) * aOuter + (long)(z % aDiv) * aInner;
  const bf16_t* Bb = Bw + (long)(z / bDiv) * bOuter + (long)(z % bDiv) * bInner;

  const int tid  = threadIdx.x;
  const int m0   = blockIdx.y * TILE_M;
  const int n0   = blockIdx.x * TILE_N;
  const int wave = tid >> 5;
  const int lane = tid & 31;
  const int r16  = lane & 15;
  const int kh   = lane >> 4;

  const bool fullM = (m0 + TILE_M) <= M;
  const bool fullN = (n0 + TILE_N) <= N;

  v8f acc[TILE_N / 16];
#pragma unroll
  for (int nt = 0; nt < TILE_N / 16; ++nt) acc[nt] = zero_v8f();

  const int AV = (TILE_M * TILE_K) / 8;  // uint4 chunks in A tile
  const int BV = (TILE_N * TILE_K) / 8;  // uint4 chunks in B tile

  for (int kk = 0; kk < K; kk += TILE_K) {
    __syncthreads();
    // ---- stage A tile (128 x 32 bf16) as 16B vectors ----
    if (fullM) {
      for (int idx = tid; idx < AV; idx += 256) {
        int r = idx >> 2, ks = (idx & 3) << 3;
        const bf16_t* gp = Ab + (long)(m0 + r) * lda + (kk + ks);
        *(uint4*)(As + r * LDSS + ks) = *(const uint4*)gp;
        __builtin_prefetch(gp + TILE_K, 0, 1);   // global_prefetch_b8: next K tile
      }
    } else {
      for (int idx = tid; idx < AV; idx += 256) {
        int r = idx >> 2, ks = (idx & 3) << 3;
        int m = m0 + r;
        uint4 v; v.x = 0u; v.y = 0u; v.z = 0u; v.w = 0u;
        if (m < M) v = *(const uint4*)(Ab + (long)m * lda + (kk + ks));
        *(uint4*)(As + r * LDSS + ks) = v;
      }
    }
    // ---- stage B tile (128 x 32 bf16), rows are output channels n ----
    if (fullN) {
      for (int idx = tid; idx < BV; idx += 256) {
        int r = idx >> 2, ks = (idx & 3) << 3;
        const bf16_t* gp = Bb + (long)(n0 + r) * ldb + (kk + ks);
        *(uint4*)(Bs + r * LDSS + ks) = *(const uint4*)gp;
        __builtin_prefetch(gp + TILE_K, 0, 1);
      }
    } else {
      for (int idx = tid; idx < BV; idx += 256) {
        int r = idx >> 2, ks = (idx & 3) << 3;
        int n = n0 + r;
        uint4 v; v.x = 0u; v.y = 0u; v.z = 0u; v.w = 0u;
        if (n < N) v = *(const uint4*)(Bb + (long)n * ldb + (kk + ks));
        *(uint4*)(Bs + r * LDSS + ks) = v;
      }
    }
    __syncthreads();

    // A fragment: lane holds row (wave*16 + lane%16);
    // K mapping per ISA 7.12.2 (16-bit A 16x32)
    v16bf afrag;
    const int ar = wave * 16 + r16;
#pragma unroll
    for (int j = 0; j < 8; ++j) {
      int k = ((j >> 2) << 4) + (kh << 3) + ((j & 3) << 1);
      afrag[2 * j]     = As[ar * LDSS + k];
      afrag[2 * j + 1] = As[ar * LDSS + k + 1];
    }
#pragma unroll
    for (int nt = 0; nt < TILE_N / 16; ++nt) {
      // B fragment: lane holds column n = nt*16 + lane%16;
      // K mapping per ISA 7.12.2 (same as A)
      v16bf bfrag;
      const int br = nt * 16 + r16;
#pragma unroll
      for (int j = 0; j < 8; ++j) {
        int k = ((j >> 2) << 4) + (kh << 3) + ((j & 3) << 1);
        bfrag[2 * j]     = Bs[br * LDSS + k];
        bfrag[2 * j + 1] = Bs[br * LDSS + k + 1];
      }
#if defined(__HIP_DEVICE_COMPILE__) && defined(__gfx1250__)
      acc[nt] = __builtin_amdgcn_wmma_f32_16x16x32_bf16(
          false, afrag, false, bfrag, (short)0, acc[nt], false, false);
#else
      (void)bfrag;
#endif
    }
  }

  const long coff = (long)(z / cDiv) * cOuter + (long)(z % cDiv) * cInner;
  float*  Cfb = Cf ? Cf + coff : nullptr;
  bf16_t* Chb = Ch ? Ch + coff : nullptr;

#pragma unroll
  for (int nt = 0; nt < TILE_N / 16; ++nt) {
    const int n = n0 + nt * 16 + r16;
    if (n >= N) continue;
    const float bv = bias ? bias[n] : 0.0f;
#pragma unroll
    for (int r = 0; r < 8; ++r) {
      const int m = m0 + wave * 16 + (kh << 3) + r;  // D: VGPR r -> M = r + 8*(lane>=16)
      if (m < M) {
        float v = acc[nt][r] * alpha + bv;
        if (Cfb) Cfb[(long)m * ldc + n] = v;
        if (Chb) Chb[(long)m * ldc + n] = (bf16_t)v;
      }
    }
  }
}

// ---------------------------------------------------------------------------
// elementwise / reduction helpers
// ---------------------------------------------------------------------------
__global__ void k_f2h(const float* __restrict__ x, bf16_t* __restrict__ y, long n) {
  long i = (long)blockIdx.x * blockDim.x + threadIdx.x;
  if (i < n) y[i] = (bf16_t)x[i];
}

// BatchNorm partial sums: grid (rowChunks, C/256); each thread owns one channel,
// accumulates 256 consecutive rows -> coalesced across threads.
__global__ __launch_bounds__(256)
void k_bn_partial(const float* __restrict__ X, float* __restrict__ psum,
                  float* __restrict__ psq, int C) {
  int rc = blockIdx.x;
  int c  = blockIdx.y * 256 + threadIdx.x;
  long r0 = (long)rc * 256;
  float s = 0.f, q = 0.f;
  for (int i = 0; i < 256; ++i) {
    float v = X[(r0 + i) * C + c];
    s += v; q += v * v;
  }
  psum[(long)rc * C + c] = s;
  psq [(long)rc * C + c] = q;
}

__global__ void k_bn_final(const float* __restrict__ psum, const float* __restrict__ psq,
                           float* __restrict__ mean, float* __restrict__ rstd,
                           int chunks, int C, float invN) {
  int c = blockIdx.x * blockDim.x + threadIdx.x;
  if (c >= C) return;
  float s = 0.f, q = 0.f;
  for (int i = 0; i < chunks; ++i) { s += psum[(long)i * C + c]; q += psq[(long)i * C + c]; }
  float m = s * invN;
  mean[c] = m;
  rstd[c] = rsqrtf(q * invN - m * m + 1e-5f);
}

__global__ void k_bn_apply(const float* __restrict__ X, const float* __restrict__ mean,
                           const float* __restrict__ rstd, const float* __restrict__ g,
                           const float* __restrict__ be,
                           float* __restrict__ Yf, bf16_t* __restrict__ Yh,
                           long total, int C) {
  long i = (long)blockIdx.x * blockDim.x + threadIdx.x;
  if (i >= total) return;
  int c = (int)(i % C);
  float y = (X[i] - mean[c]) * rstd[c] * g[c] + be[c];
  if (Yf) Yf[i] = y;
  if (Yh) Yh[i] = (bf16_t)y;
}

// apply per-(b,token) query mask and convert to bf16; tokens are i>>10 (VD=1024)
__global__ void k_mask_q(const float* __restrict__ q, const float* __restrict__ mask,
                         bf16_t* __restrict__ out, long total) {
  long i = (long)blockIdx.x * blockDim.x + threadIdx.x;
  if (i >= total) return;
  out[i] = (bf16_t)(q[i] * mask[i >> 10]);
}

// row softmax, D <= 1024, 256 threads/row
__global__ __launch_bounds__(256)
void k_softmax_row(const float* __restrict__ X, float* __restrict__ Y, int D) {
  long row = blockIdx.x;
  const float* x = X + row * D;
  float*       y = Y + row * D;
  int t = threadIdx.x;
  float buf[4];
  float mx = -3.0e38f;
#pragma unroll
  for (int j = 0; j < 4; ++j) {
    int i = t + j * 256;
    buf[j] = (i < D) ? x[i] : -3.0e38f;
    mx = fmaxf(mx, buf[j]);
  }
  __shared__ float red[256];
  red[t] = mx; __syncthreads();
  for (int s = 128; s > 0; s >>= 1) { if (t < s) red[t] = fmaxf(red[t], red[t + s]); __syncthreads(); }
  mx = red[0]; __syncthreads();
  float sum = 0.f;
#pragma unroll
  for (int j = 0; j < 4; ++j) { buf[j] = __expf(buf[j] - mx); sum += buf[j]; }
  red[t] = sum; __syncthreads();
  for (int s = 128; s > 0; s >>= 1) { if (t < s) red[t] += red[t + s]; __syncthreads(); }
  float inv = 1.0f / red[0];
#pragma unroll
  for (int j = 0; j < 4; ++j) { int i = t + j * 256; if (i < D) y[i] = buf[j] * inv; }
}

// tiny vec-mat: y[n] = x . W[n,:] + b[n]  (f32, used for the single mem-token q proj)
__global__ void k_vecmat(const float* __restrict__ x, const float* __restrict__ W,
                         const float* __restrict__ b, float* __restrict__ y, int N, int K) {
  int n = blockIdx.x * blockDim.x + threadIdx.x;
  if (n >= N) return;
  float s = b ? b[n] : 0.f;
  const float* w = W + (long)n * K;
  for (int k = 0; k < K; ++k) s += x[k] * w[k];
  y[n] = s;
}

// memory-fuse attention: 1 query (shared across batch), 64 masked keys, per (b,h)
__global__ __launch_bounds__(128)
void k_mem_attn(const float* __restrict__ q, const float* __restrict__ kf,
                const float* __restrict__ vf, const float* __restrict__ mask,
                float* __restrict__ att_out, bf16_t* __restrict__ memo) {
  int bh = blockIdx.x;
  int b = bh >> 3, h = bh & 7;
  int t = threadIdx.x;
  __shared__ float lg[64];
  __shared__ float aw[64];
  if (t < 64) {
    const float* kr = kf + ((long)(b * 64 + t)) * 1024 + h * 128;
    const float* qh = q + h * 128;
    float s = 0.f;
    for (int d = 0; d < 128; ++d) s += qh[d] * kr[d];
    lg[t] = s * mask[b * 64 + t] * 0.08838834764831845f;  // 1/sqrt(128)
  }
  __syncthreads();
  if (t == 0) {
    float mx = lg[0];
    for (int j = 1; j < 64; ++j) mx = fmaxf(mx, lg[j]);
    float s = 0.f;
    for (int j = 0; j < 64; ++j) { aw[j] = __expf(lg[j] - mx); s += aw[j]; }
    float inv = 1.f / s;
    for (int j = 0; j < 64; ++j) aw[j] *= inv;
  }
  __syncthreads();
  if (t < 64) att_out[(long)bh * 64 + t] = aw[t];
  float o = 0.f;
  for (int j = 0; j < 64; ++j)
    o += aw[j] * vf[((long)(b * 64 + j)) * 1024 + h * 128 + t] * mask[b * 64 + j];
  memo[b * 1024 + h * 128 + t] = (bf16_t)o;
}

// build cur = InstanceNorm_{L=2}(concat(neg_tok, mem_out)); write f32 out + bf16 ws
__global__ void k_cur_inorm(const float* __restrict__ negtok, const float* __restrict__ memout,
                            float* __restrict__ curf, bf16_t* __restrict__ curh) {
  int i = blockIdx.x * blockDim.x + threadIdx.x;
  if (i >= 32 * 1024) return;
  int b = i >> 10, c = i & 1023;
  float x0 = negtok[c];
  float x1 = memout[i];
  float m = 0.5f * (x0 + x1);
  float d0 = x0 - m, d1 = x1 - m;
  float rs = rsqrtf(0.5f * (d0 * d0 + d1 * d1) + 1e-5f);
  float y0 = d0 * rs, y1 = d1 * rs;
  curf[(long)b * 2048 + c]        = y0;
  curf[(long)b * 2048 + 1024 + c] = y1;
  curh[(long)b * 2048 + c]        = (bf16_t)y0;
  curh[(long)b * 2048 + 1024 + c] = (bf16_t)y1;
}

// feature-fuse attention: per (b,qi,h), 2 keys; fused softmax + A@V; write att + bf16 out
__global__ __launch_bounds__(128)
void k_ff_attn(const bf16_t* __restrict__ qff, const float* __restrict__ kff,
               const float* __restrict__ vff, float* __restrict__ attout,
               bf16_t* __restrict__ ao) {
  int id = blockIdx.x;           // (b*1024+qi)*8 + h
  int h = id & 7;
  int tok = id >> 3;             // b*1024 + qi
  int b = tok >> 10, qi = tok & 1023;
  int t = threadIdx.x;
  __shared__ float s0[128], s1[128];
  float qv = (float)qff[(long)tok * 1024 + h * 128 + t];
  float k0 = kff[(long)(b * 2) * 1024 + h * 128 + t];
  float k1 = kff[(long)(b * 2 + 1) * 1024 + h * 128 + t];
  s0[t] = qv * k0;
  s1[t] = qv * k1;
  __syncthreads();
  for (int s = 64; s > 0; s >>= 1) {
    if (t < s) { s0[t] += s0[t + s]; s1[t] += s1[t + s]; }
    __syncthreads();
  }
  __shared__ float aa0, aa1;
  if (t == 0) {
    float l0 = s0[0] * 0.08838834764831845f;
    float l1 = s1[0] * 0.08838834764831845f;
    float mx = fmaxf(l0, l1);
    float e0 = __expf(l0 - mx), e1 = __expf(l1 - mx);
    float inv = 1.f / (e0 + e1);
    aa0 = e0 * inv; aa1 = e1 * inv;
  }
  __syncthreads();
  float v0 = vff[(long)(b * 2) * 1024 + h * 128 + t];
  float v1 = vff[(long)(b * 2 + 1) * 1024 + h * 128 + t];
  ao[(long)tok * 1024 + h * 128 + t] = (bf16_t)(aa0 * v0 + aa1 * v1);
  if (t < 2) attout[((long)(b * 8 + h) * 1024 + qi) * 2 + t] = (t == 0 ? aa0 : aa1);
}

// InstanceNorm over L per (b,c): grid (B, C/256), one channel per thread (coalesced)
__global__ __launch_bounds__(256)
void k_inorm_L(const float* __restrict__ X, bf16_t* __restrict__ Y, int L, int C) {
  int b = blockIdx.x;
  int c = blockIdx.y * 256 + threadIdx.x;
  const float* xb = X + (long)b * L * C + c;
  float s = 0.f, q = 0.f;
  for (int l = 0; l < L; ++l) { float v = xb[(long)l * C]; s += v; q += v * v; }
  float invL = 1.0f / (float)L;
  float m = s * invL;
  float rs = rsqrtf(q * invL - m * m + 1e-5f);
  bf16_t* yb = Y + (long)b * L * C + c;
  for (int l = 0; l < L; ++l) yb[(long)l * C] = (bf16_t)((xb[(long)l * C] - m) * rs);
}

// ---------------------------------------------------------------------------
// launcher
// ---------------------------------------------------------------------------
extern "C" void kernel_launch(void* const* d_in, const int* in_sizes, int n_in,
                              void* d_out, int out_size, void* d_ws, size_t ws_size,
                              hipStream_t stream) {
  (void)in_sizes; (void)n_in; (void)out_size; (void)ws_size;

  const float* src    = (const float*)d_in[0];
  const float* lan    = (const float*)d_in[1];
  const float* mask   = (const float*)d_in[3];
  const float* ip_w   = (const float*)d_in[4];
  const float* ip_b   = (const float*)d_in[5];
  const float* ip_g   = (const float*)d_in[6];
  const float* ip_be  = (const float*)d_in[7];
  const float* lp_w   = (const float*)d_in[8];
  const float* lp_b   = (const float*)d_in[9];
  const float* mem_tok= (const float*)d_in[10];
  const float* neg_tok= (const float*)d_in[11];
  const float* vl_qw  = (const float*)d_in[12]; const float* vl_qb = (const float*)d_in[13];
  const float* vl_kw  = (const float*)d_in[14]; const float* vl_kb = (const float*)d_in[15];
  const float* mf_qw  = (const float*)d_in[20]; const float* mf_qb = (const float*)d_in[21];
  const float* mf_kw  = (const float*)d_in[22]; const float* mf_kb = (const float*)d_in[23];
  const float* mf_vw  = (const float*)d_in[24]; const float* mf_vb = (const float*)d_in[25];
  const float* mf_ow  = (const float*)d_in[26]; const float* mf_ob = (const float*)d_in[27];
  const float* ff_qw  = (const float*)d_in[28]; const float* ff_qb = (const float*)d_in[29];
  const float* ff_kw  = (const float*)d_in[30]; const float* ff_kb = (const float*)d_in[31];
  const float* ff_vw  = (const float*)d_in[32]; const float* ff_vb = (const float*)d_in[33];
  const float* ff_ow  = (const float*)d_in[34]; const float* ff_ob = (const float*)d_in[35];
  const float* op_w   = (const float*)d_in[36]; const float* op_b  = (const float*)d_in[37];
  const float* op_g   = (const float*)d_in[38]; const float* op_be = (const float*)d_in[39];

  const int VD = 1024, LD = 768;
  const long NTV = 32768;   // B*HW
  const long NTL = 2048;    // B*NL
  const float ALPHA_ATT = 0.08838834764831845f;  // 1/sqrt(128)

  // ---- workspace layout ----
  char* ws = (char*)d_ws;
  size_t off = 0;
  auto alloc = [&](size_t bytes) -> char* {
    char* p = ws + off;
    off += (bytes + 255) & ~(size_t)255;
    return p;
  };
  bf16_t* w_ip  = (bf16_t*)alloc((size_t)VD * VD * 2);
  bf16_t* w_lp  = (bf16_t*)alloc((size_t)VD * LD * 2);
  bf16_t* w_vlq = (bf16_t*)alloc((size_t)VD * VD * 2);
  bf16_t* w_vlk = (bf16_t*)alloc((size_t)VD * VD * 2);
  bf16_t* w_mfk = (bf16_t*)alloc((size_t)VD * VD * 2);
  bf16_t* w_mfv = (bf16_t*)alloc((size_t)VD * VD * 2);
  bf16_t* w_mfo = (bf16_t*)alloc((size_t)VD * VD * 2);
  bf16_t* w_ffq = (bf16_t*)alloc((size_t)VD * VD * 2);
  bf16_t* w_ffk = (bf16_t*)alloc((size_t)VD * VD * 2);
  bf16_t* w_ffv = (bf16_t*)alloc((size_t)VD * VD * 2);
  bf16_t* w_ffo = (bf16_t*)alloc((size_t)VD * VD * 2);
  bf16_t* w_op  = (bf16_t*)alloc((size_t)VD * VD * 2);
  bf16_t* bigHA = (bf16_t*)alloc((size_t)NTV * VD * 2);  // src_bf -> k_vl -> q_ff -> ff_n
  bf16_t* bigHB = (bf16_t*)alloc((size_t)NTV * VD * 2);  // src_p  -> attn_out
  float*  bigF  = (float*) alloc((size_t)NTV * VD * 4);  // t0 -> logits -> ff_out -> t1
  bf16_t* lan_h = (bf16_t*)alloc((size_t)NTL * LD * 2);
  bf16_t* lanp_h= (bf16_t*)alloc((size_t)NTL * VD * 2);
  float*  qvl_f = (float*) alloc((size_t)NTL * VD * 4);
  bf16_t* qvl_h = (bf16_t*)alloc((size_t)NTL * VD * 2);
  float*  kmf_f = (float*) alloc((size_t)NTL * VD * 4);
  float*  vmf_f = (float*) alloc((size_t)NTL * VD * 4);
  float*  qmf_f = (float*) alloc((size_t)VD * 4);
  bf16_t* memo_h= (bf16_t*)alloc((size_t)32 * VD * 2);
  float*  memout_f = (float*)alloc((size_t)32 * VD * 4);
  bf16_t* cur_h = (bf16_t*)alloc((size_t)64 * VD * 2);
  float*  kff_f = (float*) alloc((size_t)64 * VD * 4);
  float*  vff_f = (float*) alloc((size_t)64 * VD * 4);
  float*  psum  = (float*) alloc((size_t)128 * VD * 4);
  float*  psq   = (float*) alloc((size_t)128 * VD * 4);
  float*  meanb = (float*) alloc((size_t)VD * 4);
  float*  rstdb = (float*) alloc((size_t)VD * 4);

  // ---- output regions (flat, return order) ----
  float* out_final = (float*)d_out;                // [32,1024,1024]
  float* out_cur   = out_final + 33554432;         // [32,2,1024]
  float* out_vlatt = out_cur + 65536;              // [32,8,64,1024]
  float* out_matt  = out_vlatt + 16777216;         // [32,8,1,64]
  float* out_fatt  = out_matt + 16384;             // [32,8,1024,2]

  auto cvt = [&](const float* x, bf16_t* y, long n) {
    k_f2h<<<dim3((unsigned)((n + 255) / 256)), dim3(256), 0, stream>>>(x, y, n);
  };
  auto gemm = [&](const bf16_t* A, int lda, int aDiv, long aO, long aI,
                  const bf16_t* Bw, int ldb, int bDiv, long bO, long bI,
                  const float* bias, float* Cf, bf16_t* Ch,
                  int ldc, int cDiv, long cO, long cI,
                  int M, int N, int K, int batch, float alpha) {
    dim3 grid((N + TILE_N - 1) / TILE_N, (M + TILE_M - 1) / TILE_M, batch);
    k_gemm_bf16<<<grid, dim3(256), 0, stream>>>(A, lda, aDiv, aO, aI, Bw, ldb, bDiv, bO, bI,
                                                bias, Cf, Ch, ldc, cDiv, cO, cI, M, N, K, alpha);
  };
  auto bnstats = [&](const float* X) {
    k_bn_partial<<<dim3(128, 4), dim3(256), 0, stream>>>(X, psum, psq, VD);
    k_bn_final<<<dim3(4), dim3(256), 0, stream>>>(psum, psq, meanb, rstdb, 128, VD, 1.0f / 32768.0f);
  };

  // ---- 0) bf16 conversions ----
  cvt(src, bigHA, NTV * VD);
  cvt(lan, lan_h, NTL * LD);
  cvt(ip_w, w_ip, (long)VD * VD);   cvt(lp_w, w_lp, (long)VD * LD);
  cvt(vl_qw, w_vlq, (long)VD * VD); cvt(vl_kw, w_vlk, (long)VD * VD);
  cvt(mf_kw, w_mfk, (long)VD * VD); cvt(mf_vw, w_mfv, (long)VD * VD);
  cvt(mf_ow, w_mfo, (long)VD * VD);
  cvt(ff_qw, w_ffq, (long)VD * VD); cvt(ff_kw, w_ffk, (long)VD * VD);
  cvt(ff_vw, w_ffv, (long)VD * VD); cvt(ff_ow, w_ffo, (long)VD * VD);
  cvt(op_w, w_op, (long)VD * VD);

  // ---- 1) src_p = BN(src @ ip_w^T + ip_b) ----
  gemm(bigHA, VD, 1, 0, 0, w_ip, VD, 1, 0, 0, ip_b, bigF, nullptr, VD, 1, 0, 0,
       (int)NTV, VD, VD, 1, 1.0f);
  bnstats(bigF);
  k_bn_apply<<<dim3((unsigned)((NTV * VD + 255) / 256)), dim3(256), 0, stream>>>(
      bigF, meanb, rstdb, ip_g, ip_be, nullptr, bigHB, NTV * VD, VD);

  // ---- 2) lan_p = lan @ lp_w^T + lp_b ----
  gemm(lan_h, LD, 1, 0, 0, w_lp, LD, 1, 0, 0, lp_b, nullptr, lanp_h, VD, 1, 0, 0,
       (int)NTL, VD, LD, 1, 1.0f);

  // ---- 3) vl attention logits + softmax (vl_out is dead code) ----
  gemm(lanp_h, VD, 1, 0, 0, w_vlq, VD, 1, 0, 0, vl_qb, qvl_f, nullptr, VD, 1, 0, 0,
       (int)NTL, VD, VD, 1, 1.0f);
  k_mask_q<<<dim3((unsigned)((NTL * VD + 255) / 256)), dim3(256), 0, stream>>>(
      qvl_f, mask, qvl_h, NTL * VD);
  gemm(bigHB, VD, 1, 0, 0, w_vlk, VD, 1, 0, 0, vl_kb, nullptr, bigHA, VD, 1, 0, 0,
       (int)NTV, VD, VD, 1, 1.0f);
  // logits[z=b*8+h]: q[b,:,h*128:] @ k[b,:,h*128:]^T / sqrt(128)
  gemm(qvl_h, VD, 8, 65536, 128,
       bigHA, VD, 8, 1048576, 128,
       nullptr, bigF, nullptr, 1024, 1, 65536, 0,
       64, 1024, 128, 256, ALPHA_ATT);
  k_softmax_row<<<dim3(16384), dim3(256), 0, stream>>>(bigF, out_vlatt, 1024);

  // ---- 4) memory fuse ----
  gemm(lanp_h, VD, 1, 0, 0, w_mfk, VD, 1, 0, 0, mf_kb, kmf_f, nullptr, VD, 1, 0, 0,
       (int)NTL, VD, VD, 1, 1.0f);
  gemm(lanp_h, VD, 1, 0, 0, w_mfv, VD, 1, 0, 0, mf_vb, vmf_f, nullptr, VD, 1, 0, 0,
       (int)NTL, VD, VD, 1, 1.0f);
  k_vecmat<<<dim3(4), dim3(256), 0, stream>>>(mem_tok, mf_qw, mf_qb, qmf_f, VD, VD);
  k_mem_attn<<<dim3(256), dim3(128), 0, stream>>>(qmf_f, kmf_f, vmf_f, mask, out_matt, memo_h);
  gemm(memo_h, VD, 1, 0, 0, w_mfo, VD, 1, 0, 0, mf_ob, memout_f, nullptr, VD, 1, 0, 0,
       32, VD, VD, 1, 1.0f);
  k_cur_inorm<<<dim3(128), dim3(256), 0, stream>>>(neg_tok, memout_f, out_cur, cur_h);

  // ---- 5) feature fuse ----
  gemm(bigHB, VD, 1, 0, 0, w_ffq, VD, 1, 0, 0, ff_qb, nullptr, bigHA, VD, 1, 0, 0,
       (int)NTV, VD, VD, 1, 1.0f);
  gemm(cur_h, VD, 1, 0, 0, w_ffk, VD, 1, 0, 0, ff_kb, kff_f, nullptr, VD, 1, 0, 0,
       64, VD, VD, 1, 1.0f);
  gemm(cur_h, VD, 1, 0, 0, w_ffv, VD, 1, 0, 0, ff_vb, vff_f, nullptr, VD, 1, 0, 0,
       64, VD, VD, 1, 1.0f);
  k_ff_attn<<<dim3(262144), dim3(128), 0, stream>>>(bigHA, kff_f, vff_f, out_fatt, bigHB);
  gemm(bigHB, VD, 1, 0, 0, w_ffo, VD, 1, 0, 0, ff_ob, bigF, nullptr, VD, 1, 0, 0,
       (int)NTV, VD, VD, 1, 1.0f);
  k_inorm_L<<<dim3(32, 4), dim3(256), 0, stream>>>(bigF, bigHA, 1024, VD);

  // ---- 6) final = BN(ff_n @ op_w^T + op_b) ----
  gemm(bigHA, VD, 1, 0, 0, w_op, VD, 1, 0, 0, op_b, bigF, nullptr, VD, 1, 0, 0,
       (int)NTV, VD, VD, 1, 1.0f);
  bnstats(bigF);
  k_bn_apply<<<dim3((unsigned)((NTV * VD + 255) / 256)), dim3(256), 0, stream>>>(
      bigF, meanb, rstdb, op_g, op_be, out_final, nullptr, NTV * VD, VD);
}
